// WindowMCA_26749056319778
// MI455X (gfx1250) — matrix operand order repaired and checked by reading, not compiled, not run on hardware
//
#include <hip/hip_runtime.h>
#include <hip/hip_bf16.h>

typedef __attribute__((ext_vector_type(16))) _Float16 v16h;
typedef __attribute__((ext_vector_type(8)))  float    v8f;

#define NTOK 49
#define NPAD 64
#define CDIM 128
#define HEADS 4
#define HD 32
#define NWIN 64

// ---------- pre-kernels ----------

__global__ void wmca_convert_weights(const float* __restrict__ Wq,
                                     const float* __restrict__ Wk,
                                     const float* __restrict__ Wv,
                                     const float* __restrict__ Wo,
                                     _Float16* __restrict__ dst) {
  int i = blockIdx.x * blockDim.x + threadIdx.x;
  if (i >= 4 * 16384) return;
  const float* src = (i < 16384) ? Wq : ((i < 32768) ? Wk : ((i < 49152) ? Wv : Wo));
  dst[i] = (_Float16)src[i & 16383];
}

// biasPad[h][64][64]: rel-pos bias with -1e30 on padded key columns
__global__ void wmca_bias_pad(const float* __restrict__ rpb,
                              const int* __restrict__ rel,
                              float* __restrict__ dst) {
  int i = blockIdx.x * blockDim.x + threadIdx.x;
  if (i >= HEADS * NPAD * NPAD) return;
  int h = i >> 12, rem = i & 4095, q = rem >> 6, k = rem & 63;
  float val;
  if (k >= NTOK)      val = -1.0e30f;
  else if (q >= NTOK) val = 0.0f;
  else                val = rpb[rel[q * NTOK + k] * HEADS + h];
  dst[i] = val;
}

// ---------- fragment helpers ----------

// A-fragment (16x32 f16) from row-major f32 global, zeroing padded token rows.
__device__ __forceinline__ v16h aFragF32(const float* __restrict__ xw,
                                         int tok, int k0, int g) {
  v16h a;
  if (tok < NTOK) {
    const float* p0 = xw + tok * CDIM + k0 + 16 * g;  // K = k0+16g .. +7
    const float* p1 = p0 + 8;                         // K = k0+16g+8 .. +7
#pragma unroll
    for (int i = 0; i < 8; ++i) { a[i] = (_Float16)p0[i]; a[i + 8] = (_Float16)p1[i]; }
  } else {
#pragma unroll
    for (int i = 0; i < 16; ++i) a[i] = (_Float16)0.0f;
  }
  return a;
}

// A-fragment from row-major f16 LDS.
__device__ __forceinline__ v16h aFragLds(const _Float16* X, int ld,
                                         int row, int k0, int g) {
  const _Float16* p0 = X + row * ld + k0 + 16 * g;
  v16h a;
#pragma unroll
  for (int i = 0; i < 8; ++i) { a[i] = p0[i]; a[i + 8] = p0[8 + i]; }
  return a;
}

#define WMMA_F16(a, b, c) \
  __builtin_amdgcn_wmma_f32_16x16x32_f16(false, (a), false, (b), (short)0, (c), false, false)

// ---------- fused window attention ----------

__global__ __launch_bounds__(256) void wmca_main(
    const float* __restrict__ q_in, const float* __restrict__ k_in,
    const float* __restrict__ v_in, const float* __restrict__ mask,
    const float* __restrict__ bq, const float* __restrict__ bk,
    const float* __restrict__ bv, const float* __restrict__ bo,
    const _Float16* __restrict__ wf16,   // [4][128*128] f16: Wq,Wk,Wv,Wo
    const float* __restrict__ biasPad,   // [4][64][64]
    float* __restrict__ out) {
  __shared__ __align__(128) _Float16 smem[32768];       // 64 KB
  _Float16* Qs = smem;                                  // [64][128] scaled Q
  _Float16* Ks = smem + 8192;                           // [64][128]
  _Float16* Ps = smem;                                  // [4][64][64] reuses Q+K
  _Float16* Vt = smem + 16384;                          // [128][64] V transposed
  _Float16* Xs = smem + 24576;                          // [64][128] attn output

  const int b    = blockIdx.x;
  const int tid  = threadIdx.x;
  const int w    = tid >> 5;
  const int lane = tid & 31;
  const int g    = lane >> 4;
  const int ln   = lane & 15;
  const float scale = 0.17677669529663687f;             // 32^-0.5

  const float* xq = q_in + (size_t)b * NTOK * CDIM;
  const float* xk = k_in + (size_t)b * NTOK * CDIM;
  const float* xv = v_in + (size_t)b * NTOK * CDIM;

  // ---- Phase A: Q/K/V projections (96 tiles, 12 per wave) ----
  for (int t = 0; t < 12; ++t) {
    int T = w * 12 + t;
    int p = T >> 5;                 // 0=Q, 1=K, 2=V
    int rem = T & 31;
    int m = rem >> 3, n = rem & 7;
    const float* xp = (p == 0) ? xq : ((p == 1) ? xk : xv);
    const float* bp = (p == 0) ? bq : ((p == 1) ? bk : bv);
    const _Float16* Wp = wf16 + p * 16384;
    v8f acc = {};
#pragma unroll
    for (int kk = 0; kk < 4; ++kk) {
      int k0 = kk * 32;
      v16h a   = aFragF32(xp, m * 16 + ln, k0, g);
      v16h bfr = *(const v16h*)(Wp + (n * 16 + ln) * CDIM + k0 + 16 * g);
      acc = WMMA_F16(a, bfr, acc);
    }
    int col = n * 16 + ln;
    float bias = bp[col];
#pragma unroll
    for (int v = 0; v < 8; ++v) {
      int row = m * 16 + v + 8 * g;
      float val = acc[v] + bias;
      if (p == 0)      Qs[row * CDIM + col] = (_Float16)(val * scale);
      else if (p == 1) Ks[row * CDIM + col] = (_Float16)val;
      else             Vt[col * NPAD + row] = (_Float16)val;
    }
  }
  __syncthreads();

  // ---- Phase B: S = Q K^T (+bias+mask), softmax -> P ----
  // wave owns head h, query rows [32*mh, 32*mh+32) => complete rows.
  const int h  = w >> 1;
  const int mh = w & 1;
  const float* bmh = biasPad + h * (NPAD * NPAD);
  const float* mw  = mask + (size_t)(b & (NWIN - 1)) * NTOK * NTOK;

  v8f sacc[2][4];
#pragma unroll
  for (int mi = 0; mi < 2; ++mi) {
    int mt = mh * 2 + mi;
    v16h a = aFragLds(Qs, CDIM, mt * 16 + ln, h * HD, g);
#pragma unroll
    for (int nt = 0; nt < 4; ++nt) {
      v16h bfr = *(const v16h*)(Ks + (nt * 16 + ln) * CDIM + h * HD + 16 * g);
      v8f acc = {};
      sacc[mi][nt] = WMMA_F16(a, bfr, acc);
    }
  }

  // add relative-position bias (with -inf key padding) + shifted-window mask
#pragma unroll
  for (int mi = 0; mi < 2; ++mi) {
    int mt = mh * 2 + mi;
#pragma unroll
    for (int nt = 0; nt < 4; ++nt) {
      int k = nt * 16 + ln;
#pragma unroll
      for (int v = 0; v < 8; ++v) {
        int q = mt * 16 + v + 8 * g;
        float add = bmh[q * NPAD + k];
        if (q < NTOK && k < NTOK) add += mw[q * NTOK + k];
        sacc[mi][nt][v] += add;
      }
    }
  }

  __syncthreads();   // all waves done reading Qs/Ks; safe to overwrite with P

  // softmax per row: 16-lane shuffle reductions inside each half-wave
#pragma unroll
  for (int mi = 0; mi < 2; ++mi) {
    int mt = mh * 2 + mi;
#pragma unroll
    for (int v = 0; v < 8; ++v) {
      float mx = -3.0e38f;
#pragma unroll
      for (int nt = 0; nt < 4; ++nt) mx = fmaxf(mx, sacc[mi][nt][v]);
#pragma unroll
      for (int off = 1; off < 16; off <<= 1) mx = fmaxf(mx, __shfl_xor(mx, off, 32));
      float e[4];
      float sum = 0.0f;
#pragma unroll
      for (int nt = 0; nt < 4; ++nt) { e[nt] = __expf(sacc[mi][nt][v] - mx); sum += e[nt]; }
#pragma unroll
      for (int off = 1; off < 16; off <<= 1) sum += __shfl_xor(sum, off, 32);
      float inv = 1.0f / sum;
      int q = mt * 16 + v + 8 * g;
#pragma unroll
      for (int nt = 0; nt < 4; ++nt)
        Ps[h * 4096 + q * NPAD + nt * 16 + ln] = (_Float16)(e[nt] * inv);
    }
  }
  __syncthreads();

  // ---- Phase C: O = P V  (per head, K=64 => 2 WMMA steps) ----
#pragma unroll
  for (int mi = 0; mi < 2; ++mi) {
    int mt = mh * 2 + mi;
    v8f acc0 = {}, acc1 = {};
#pragma unroll
    for (int kk = 0; kk < 2; ++kk) {
      int k0 = kk * 32;
      v16h a  = aFragLds(Ps + h * 4096, NPAD, mt * 16 + ln, k0, g);
      v16h b0 = *(const v16h*)(Vt + (h * HD + 0 * 16 + ln) * NPAD + k0 + 16 * g);
      v16h b1 = *(const v16h*)(Vt + (h * HD + 1 * 16 + ln) * NPAD + k0 + 16 * g);
      acc0 = WMMA_F16(a, b0, acc0);
      acc1 = WMMA_F16(a, b1, acc1);
    }
#pragma unroll
    for (int nt = 0; nt < 2; ++nt) {
      int col = h * HD + nt * 16 + ln;
#pragma unroll
      for (int v = 0; v < 8; ++v) {
        int row = mt * 16 + v + 8 * g;
        float val = (nt == 0) ? acc0[v] : acc1[v];
        Xs[row * CDIM + col] = (_Float16)val;
      }
    }
  }
  __syncthreads();

  // ---- Phase D: out projection + bias, f32 store (valid rows only) ----
  const _Float16* Wo16 = wf16 + 3 * 16384;
  float* outw = out + (size_t)b * NTOK * CDIM;
  for (int t = 0; t < 4; ++t) {
    int T = w * 4 + t;
    int m = T >> 3, n = T & 7;
    v8f acc = {};
#pragma unroll
    for (int kk = 0; kk < 4; ++kk) {
      int k0 = kk * 32;
      v16h a   = aFragLds(Xs, CDIM, m * 16 + ln, k0, g);
      v16h bfr = *(const v16h*)(Wo16 + (n * 16 + ln) * CDIM + k0 + 16 * g);
      acc = WMMA_F16(a, bfr, acc);
    }
    int col = n * 16 + ln;
    float bias = bo[col];
#pragma unroll
    for (int v = 0; v < 8; ++v) {
      int row = m * 16 + v + 8 * g;
      if (row < NTOK) outw[row * CDIM + col] = acc[v] + bias;
    }
  }
}

// ---------- launch ----------

extern "C" void kernel_launch(void* const* d_in, const int* in_sizes, int n_in,
                              void* d_out, int out_size, void* d_ws, size_t ws_size,
                              hipStream_t stream) {
  (void)in_sizes; (void)n_in; (void)out_size; (void)ws_size;
  const float* q_in = (const float*)d_in[0];
  const float* k_in = (const float*)d_in[1];
  const float* v_in = (const float*)d_in[2];
  const float* mask = (const float*)d_in[3];
  const float* Wq   = (const float*)d_in[4];
  const float* bq   = (const float*)d_in[5];
  const float* Wk   = (const float*)d_in[6];
  const float* bk   = (const float*)d_in[7];
  const float* Wv   = (const float*)d_in[8];
  const float* bv   = (const float*)d_in[9];
  const float* Wo   = (const float*)d_in[10];
  const float* bo   = (const float*)d_in[11];
  const float* rpb  = (const float*)d_in[12];
  const int*   rel  = (const int*)d_in[13];
  float* out = (float*)d_out;

  _Float16* wf16 = (_Float16*)d_ws;                       // 131072 B
  float* biasPad = (float*)((char*)d_ws + 131072);        // 65536 B

  wmca_convert_weights<<<256, 256, 0, stream>>>(Wq, Wk, Wv, Wo, wf16);
  wmca_bias_pad<<<64, 256, 0, stream>>>(rpb, rel, biasPad);
  wmca_main<<<4096, 256, 0, stream>>>(q_in, k_in, v_in, mask,
                                      bq, bk, bv, bo, wf16, biasPad, out);
}
